// TransformerBlock_63891933495742
// MI455X (gfx1250) — matrix operand order repaired and checked by reading, not compiled, not run on hardware
//
#include <hip/hip_runtime.h>
#include <hip/hip_bf16.h>
#include <stdint.h>

// ---------------- problem constants (from reference) ----------------
#define BB   2
#define TT   2048
#define EE   1024
#define HH   16
#define DHH  64
#define DFFN 4096
#define MM   (BB*TT)      // 4096 token rows

typedef __attribute__((ext_vector_type(16))) __bf16 bf16x16;
typedef __attribute__((ext_vector_type(8)))  __bf16 bf16x8;
typedef __attribute__((ext_vector_type(4)))  __bf16 bf16x4;
typedef __attribute__((ext_vector_type(8)))  float  f32x8;

__device__ __forceinline__ __bf16 f2bf(float f) {
  union { float f; uint32_t u; } in; in.f = f;
  uint32_t r = in.u + 0x7FFFu + ((in.u >> 16) & 1u);   // round-to-nearest-even
  union { uint16_t s; __bf16 b; } out; out.s = (uint16_t)(r >> 16);
  return out.b;
}

__device__ __forceinline__ bf16x16 join8(bf16x8 lo, bf16x8 hi) {
  union { bf16x16 v; bf16x8 h[2]; } u; u.h[0] = lo; u.h[1] = hi; return u.v;
}

// ---------------- LayerNorm (fp32 in -> bf16 out), one block per row ----------------
__global__ __launch_bounds__(256)
void ln_bf16_kernel(const float* __restrict__ x, const float* __restrict__ gamma,
                    const float* __restrict__ beta, __bf16* __restrict__ out) {
  __shared__ float red[256];
  const int row = blockIdx.x;
  const int t   = threadIdx.x;
  const float4 v = ((const float4*)(x + (size_t)row * EE))[t];
  red[t] = v.x + v.y + v.z + v.w;
  __syncthreads();
  for (int off = 128; off > 0; off >>= 1) {
    if (t < off) red[t] += red[t + off];
    __syncthreads();
  }
  const float mean = red[0] * (1.0f / EE);
  __syncthreads();
  const float dx0 = v.x - mean, dx1 = v.y - mean, dx2 = v.z - mean, dx3 = v.w - mean;
  red[t] = dx0*dx0 + dx1*dx1 + dx2*dx2 + dx3*dx3;
  __syncthreads();
  for (int off = 128; off > 0; off >>= 1) {
    if (t < off) red[t] += red[t + off];
    __syncthreads();
  }
  const float rstd = rsqrtf(red[0] * (1.0f / EE) + 1e-5f);
  const float4 g = ((const float4*)gamma)[t];
  const float4 b = ((const float4*)beta)[t];
  bf16x4 o;
  o[0] = f2bf(dx0 * rstd * g.x + b.x);
  o[1] = f2bf(dx1 * rstd * g.y + b.y);
  o[2] = f2bf(dx2 * rstd * g.z + b.z);
  o[3] = f2bf(dx3 * rstd * g.w + b.w);
  ((bf16x4*)(out + (size_t)row * EE))[t] = o;
}

// ---------------- packing kernels ----------------
__global__ void cast_bf16_kernel(const float* __restrict__ in, __bf16* __restrict__ out, int n) {
  int i = blockIdx.x * 256 + threadIdx.x;
  if (i < n) out[i] = f2bf(in[i]);
}

// Wq/Wk/Wv [H,E,DH] fp32  ->  Wqkv [E, 3E] bf16 with col = which*E + h*DH + d
__global__ void pack_qkv_w_kernel(const float* __restrict__ Wq, const float* __restrict__ Wk,
                                  const float* __restrict__ Wv, __bf16* __restrict__ out) {
  int i = blockIdx.x * 256 + threadIdx.x;          // < 3 * 1048576
  int which = i >> 20;
  int rem   = i & 0xFFFFF;                         // h*65536 + e*64 + d
  const float* src = (which == 0) ? Wq : (which == 1) ? Wk : Wv;
  int h = rem >> 16;
  int e = (rem >> 6) & 1023;
  int d = rem & 63;
  out[(size_t)e * (3*EE) + which*EE + h*DHH + d] = f2bf(src[rem]);
}

__global__ void pack_qkv_b_kernel(const float* __restrict__ bq, const float* __restrict__ bk,
                                  const float* __restrict__ bv, float* __restrict__ out) {
  int i = blockIdx.x * 256 + threadIdx.x;          // < 3072
  const float* src = (i < EE) ? bq : (i < 2*EE) ? bk : bv;
  out[i] = src[i & (EE - 1)];
}

// ---------------- generic bf16 GEMM, f32 accumulate on WMMA ----------------
// C = A[M,K] @ B[K,N] + bias, epilogues:
//   EPI=0: store bf16         EPI=1: +residual, store fp32        EPI=2: ReLU, store bf16
// Block tile 128x256x32, 8 waves in a 2(M) x 4(N) grid, 64x64 per wave (16 WMMA accs).
// Software pipeline: next K-tile staged in VGPRs while current tile is in the WMMAs.
#define BM  128
#define BN  256
#define BKK 32
#define LDA 40     // padded LDS strides (80B rows, 16B aligned)
#define LDB 40

template<int EPI>
__global__ __launch_bounds__(256)
void gemm_bf16_kernel(const __bf16* __restrict__ A, const __bf16* __restrict__ Bm,
                      const float* __restrict__ bias, const float* __restrict__ res,
                      __bf16* __restrict__ Cb, float* __restrict__ Cf,
                      int M, int N, int K) {
  __shared__ __align__(16) __bf16 As[BM * LDA];   // 10 KB
  __shared__ __align__(16) __bf16 Bs[BN * LDB];   // 20 KB, stored transposed: Bs[n][k]

  const int t    = threadIdx.x;
  const int lane = t & 31;
  const int wave = t >> 5;
  const int wm   = wave >> 2;          // 0..1 -> 64-row strip
  const int wn   = wave & 3;           // 0..3 -> 64-col strip
  const int lm   = lane & 15;
  const int lhi  = (lane >> 4) * 8;    // C/D row offset
  const int lk8  = (lane >> 4) * 8;    // A fragment K base
  const int lk16 = (lane >> 4) * 16;   // unused; B fragments also use lk8

  const int m0 = blockIdx.y * BM;
  const int n0 = blockIdx.x * BN;

  // per-thread staging coordinates: A = 512 chunks (2/thr), B = 1024 chunks (4/thr)
  int ar[2], ak[2], br[4], bn8[4];
#pragma unroll
  for (int c = 0; c < 2; c++) { int idx = t + c*256; ar[c] = idx >> 2; ak[c] = (idx & 3) * 8; }
#pragma unroll
  for (int c = 0; c < 4; c++) { int idx = t + c*256; br[c] = idx >> 5; bn8[c] = (idx & 31) * 8; }

  f32x8 acc[4][4];
#pragma unroll
  for (int i = 0; i < 4; i++)
#pragma unroll
    for (int j = 0; j < 4; j++)
#pragma unroll
      for (int r = 0; r < 8; r++) acc[i][j][r] = 0.0f;

  // prologue: stage K-tile 0 straight into LDS
#pragma unroll
  for (int c = 0; c < 2; c++)
    *(bf16x8*)&As[ar[c] * LDA + ak[c]] =
        *(const bf16x8*)(A + (size_t)(m0 + ar[c]) * K + ak[c]);
#pragma unroll
  for (int c = 0; c < 4; c++) {
    bf16x8 v = *(const bf16x8*)(Bm + (size_t)br[c] * N + n0 + bn8[c]);
#pragma unroll
    for (int i = 0; i < 8; i++) Bs[(bn8[c] + i) * LDB + br[c]] = v[i];
  }
  __syncthreads();

  for (int k0 = 0; k0 < K; k0 += BKK) {
    const bool more = (k0 + BKK) < K;

    // issue next tile's global loads early (latency overlapped with WMMAs)
    bf16x8 astg[2], bstg[4];
    if (more) {
#pragma unroll
      for (int c = 0; c < 2; c++)
        astg[c] = *(const bf16x8*)(A + (size_t)(m0 + ar[c]) * K + k0 + BKK + ak[c]);
#pragma unroll
      for (int c = 0; c < 4; c++)
        bstg[c] = *(const bf16x8*)(Bm + (size_t)(k0 + BKK + br[c]) * N + n0 + bn8[c]);
    }

    // compute current tile from LDS: 4 A frags, 4 B frags, 16 WMMAs
    bf16x16 af[4];
#pragma unroll
    for (int mi = 0; mi < 4; mi++) {
      const __bf16* p = &As[(wm*64 + mi*16 + lm) * LDA + lk8];
      af[mi] = join8(*(const bf16x8*)p, *(const bf16x8*)(p + 16));
    }
#pragma unroll
    for (int ni = 0; ni < 4; ni++) {
      const __bf16* p = &Bs[(wn*64 + ni*16 + lm) * LDB + lk8];
      bf16x16 bf = join8(*(const bf16x8*)p, *(const bf16x8*)(p + 16));
#pragma unroll
      for (int mi = 0; mi < 4; mi++)
        acc[mi][ni] = __builtin_amdgcn_wmma_f32_16x16x32_bf16(
            false, af[mi], false, bf, (short)0, acc[mi][ni], false, false);
    }

    __syncthreads();
    if (more) {
#pragma unroll
      for (int c = 0; c < 2; c++)
        *(bf16x8*)&As[ar[c] * LDA + ak[c]] = astg[c];
#pragma unroll
      for (int c = 0; c < 4; c++)
#pragma unroll
        for (int i = 0; i < 8; i++) Bs[(bn8[c] + i) * LDB + br[c]] = bstg[c][i];
      __syncthreads();
    }
  }

  // fused epilogue
#pragma unroll
  for (int mi = 0; mi < 4; mi++) {
#pragma unroll
    for (int ni = 0; ni < 4; ni++) {
      const int gcol = n0 + wn*64 + ni*16 + lm;
      const float bv = bias[gcol];
#pragma unroll
      for (int r = 0; r < 8; r++) {
        const int grow = m0 + wm*64 + mi*16 + r + lhi;
        float v = acc[mi][ni][r] + bv;
        if (EPI == 1) {
          Cf[(size_t)grow * N + gcol] = v + res[(size_t)grow * N + gcol];
        } else {
          if (EPI == 2) v = fmaxf(v, 0.0f);
          Cb[(size_t)grow * N + gcol] = f2bf(v);
        }
      }
    }
  }
}

// ---------------- causal flash attention ----------------
// qkv: [B*T, 3E] bf16 rows = [q | k | v] head-concatenated. O: [B*T, E] bf16.
#define QROWS 128
#define KCOLS 64
#define LQP   72

__global__ __launch_bounds__(256)
void flash_attn_kernel(const __bf16* __restrict__ qkv, __bf16* __restrict__ O) {
  __shared__ __align__(16) __bf16 QP[QROWS * LQP]; // Q tile, then reused as P tile
  __shared__ __align__(16) __bf16 Kt[KCOLS * LQP]; // [s][d]
  __shared__ __align__(16) __bf16 Vt[DHH  * LQP];  // [d][s] (transposed)

  const int t    = threadIdx.x;
  const int lane = t & 31;
  const int wave = t >> 5;        // each wave owns a 16-row query strip
  const int lm   = lane & 15;
  const int lhi  = (lane >> 4) * 8;
  const int lk8  = (lane >> 4) * 8;
  const int lk16 = (lane >> 4) * 16;
  const int mrow = wave * 16;

  const int q0 = blockIdx.x * QROWS;
  const int bh = blockIdx.y;
  const int b  = bh >> 4;
  const int h  = bh & 15;

  const size_t rs = 3 * EE;
  const __bf16* qbase = qkv + (size_t)(b*TT + q0) * rs + h*DHH;
  const __bf16* kbase = qkv + (size_t)(b*TT) * rs + EE   + h*DHH;
  const __bf16* vbase = qkv + (size_t)(b*TT) * rs + 2*EE + h*DHH;

  // per-thread K/V staging coordinates (64x64 tile -> 512 chunks, 2/thr)
  int kr[2], kc8[2];
#pragma unroll
  for (int c = 0; c < 2; c++) { int idx = t + c*256; kr[c] = idx >> 3; kc8[c] = (idx & 7) * 8; }

  // prologue: Q tile (128x64) and K/V tile 0 into LDS
#pragma unroll
  for (int c = 0; c < 4; c++) {
    int idx = t + c * 256;
    int r = idx >> 3, c8 = (idx & 7) * 8;
    *(bf16x8*)&QP[r * LQP + c8] = *(const bf16x8*)(qbase + (size_t)r * rs + c8);
  }
#pragma unroll
  for (int c = 0; c < 2; c++) {
    *(bf16x8*)&Kt[kr[c] * LQP + kc8[c]] = *(const bf16x8*)(kbase + (size_t)kr[c] * rs + kc8[c]);
    bf16x8 v = *(const bf16x8*)(vbase + (size_t)kr[c] * rs + kc8[c]);
#pragma unroll
    for (int i = 0; i < 8; i++) Vt[(kc8[c] + i) * LQP + kr[c]] = v[i];
  }
  __syncthreads();

  bf16x16 qf[2];                           // Q A-fragments held in registers
#pragma unroll
  for (int ks = 0; ks < 2; ks++) {
    const __bf16* p = &QP[(mrow + lm) * LQP + ks*32 + lk8];
    qf[ks] = join8(*(const bf16x8*)p, *(const bf16x8*)(p + 16));
  }

  float mrun[8], lrun[8];
  f32x8 oacc[4];
#pragma unroll
  for (int r = 0; r < 8; r++) { mrun[r] = -3.0e38f; lrun[r] = 0.0f; }
#pragma unroll
  for (int nt = 0; nt < 4; nt++)
#pragma unroll
    for (int r = 0; r < 8; r++) oacc[nt][r] = 0.0f;

  const int nkb = 2 * blockIdx.x + 2;      // causal: only lower-triangular key tiles
  for (int kb = 0; kb < nkb; kb++) {
    const int s0 = kb * KCOLS;
    const bool more = (kb + 1) < nkb;

    // stage next K/V tile in VGPRs while this tile is in the matrix units
    bf16x8 kstg[2], vstg[2];
    if (more) {
#pragma unroll
      for (int c = 0; c < 2; c++) {
        kstg[c] = *(const bf16x8*)(kbase + (size_t)(s0 + KCOLS + kr[c]) * rs + kc8[c]);
        vstg[c] = *(const bf16x8*)(vbase + (size_t)(s0 + KCOLS + kr[c]) * rs + kc8[c]);
      }
    }

    // S = Q @ K^T for this wave's strip (16 x 64)
    f32x8 sc[4];
#pragma unroll
    for (int nt = 0; nt < 4; nt++) {
#pragma unroll
      for (int r = 0; r < 8; r++) sc[nt][r] = 0.0f;
#pragma unroll
      for (int ks = 0; ks < 2; ks++) {
        const __bf16* p = &Kt[(nt*16 + lm) * LQP + ks*32 + lk8];
        bf16x16 kf = join8(*(const bf16x8*)p, *(const bf16x8*)(p + 16));
        sc[nt] = __builtin_amdgcn_wmma_f32_16x16x32_bf16(
            false, qf[ks], false, kf, (short)0, sc[nt], false, false);
      }
    }

    // scale + causal mask
#pragma unroll
    for (int nt = 0; nt < 4; nt++) {
      const int scol = s0 + nt*16 + lm;
#pragma unroll
      for (int r = 0; r < 8; r++) {
        const int trow = q0 + mrow + r + lhi;
        float v = sc[nt][r] * 0.125f;                 // 1/sqrt(64)
        sc[nt][r] = (scol > trow) ? -1.0e9f : v;
      }
    }

    // online softmax (row stats shared by the 16 lanes of each half-wave)
#pragma unroll
    for (int r = 0; r < 8; r++) {
      float mx = fmaxf(fmaxf(sc[0][r], sc[1][r]), fmaxf(sc[2][r], sc[3][r]));
#pragma unroll
      for (int msk = 1; msk < 16; msk <<= 1) mx = fmaxf(mx, __shfl_xor(mx, msk, 32));
      const float mnew  = fmaxf(mrun[r], mx);
      const float alpha = __expf(mrun[r] - mnew);
      float psum = 0.0f;
#pragma unroll
      for (int nt = 0; nt < 4; nt++) {
        float p = __expf(sc[nt][r] - mnew);
        sc[nt][r] = p;
        psum += p;
      }
#pragma unroll
      for (int msk = 1; msk < 16; msk <<= 1) psum += __shfl_xor(psum, msk, 32);
      lrun[r] = lrun[r] * alpha + psum;
      mrun[r] = mnew;
#pragma unroll
      for (int nt = 0; nt < 4; nt++) oacc[nt][r] *= alpha;
    }

    // transpose P through LDS (own strip only; per-wave DS is in-order), then O += P @ V
#pragma unroll
    for (int nt = 0; nt < 4; nt++)
#pragma unroll
      for (int r = 0; r < 8; r++)
        QP[(mrow + r + lhi) * LQP + nt*16 + lm] = f2bf(sc[nt][r]);

#pragma unroll
    for (int ks = 0; ks < 2; ks++) {
      const __bf16* pp = &QP[(mrow + lm) * LQP + ks*32 + lk8];
      bf16x16 pf = join8(*(const bf16x8*)pp, *(const bf16x8*)(pp + 16));
#pragma unroll
      for (int nt2 = 0; nt2 < 4; nt2++) {
        const __bf16* pv = &Vt[(nt2*16 + lm) * LQP + ks*32 + lk8];
        bf16x16 vf = join8(*(const bf16x8*)pv, *(const bf16x8*)(pv + 16));
        oacc[nt2] = __builtin_amdgcn_wmma_f32_16x16x32_bf16(
            false, pf, false, vf, (short)0, oacc[nt2], false, false);
      }
    }

    __syncthreads();
    if (more) {
#pragma unroll
      for (int c = 0; c < 2; c++) {
        *(bf16x8*)&Kt[kr[c] * LQP + kc8[c]] = kstg[c];
#pragma unroll
        for (int i = 0; i < 8; i++) Vt[(kc8[c] + i) * LQP + kr[c]] = vstg[c][i];
      }
      __syncthreads();
    }
  }

  // normalize + store head-concatenated output
  __bf16* obase = O + (size_t)(b*TT + q0 + mrow) * EE + h*DHH;
#pragma unroll
  for (int r = 0; r < 8; r++) {
    const float inv = 1.0f / lrun[r];
#pragma unroll
    for (int nt2 = 0; nt2 < 4; nt2++)
      obase[(size_t)(r + lhi) * EE + nt2*16 + lm] = f2bf(oacc[nt2][r] * inv);
  }
}

// ---------------- host orchestration ----------------
extern "C" void kernel_launch(void* const* d_in, const int* in_sizes, int n_in,
                              void* d_out, int out_size, void* d_ws, size_t ws_size,
                              hipStream_t stream) {
  (void)in_sizes; (void)n_in; (void)out_size; (void)ws_size;
  const float* x   = (const float*)d_in[0];
  const float* Wq  = (const float*)d_in[1];
  const float* bq  = (const float*)d_in[2];
  const float* Wk  = (const float*)d_in[3];
  const float* bk  = (const float*)d_in[4];
  const float* Wv  = (const float*)d_in[5];
  const float* bv  = (const float*)d_in[6];
  const float* Wo  = (const float*)d_in[7];
  const float* bo  = (const float*)d_in[8];
  const float* W1  = (const float*)d_in[9];
  const float* b1  = (const float*)d_in[10];
  const float* W2  = (const float*)d_in[11];
  const float* b2  = (const float*)d_in[12];
  const float* g1  = (const float*)d_in[13];
  const float* be1 = (const float*)d_in[14];
  const float* g2  = (const float*)d_in[15];
  const float* be2 = (const float*)d_in[16];
  float* out = (float*)d_out;

  // workspace carve (~80 MiB total)
  char* ws = (char*)d_ws;
  size_t cur = 0;
  auto carve = [&](size_t bytes) -> char* {
    char* p = ws + cur;
    cur += (bytes + 255) & ~(size_t)255;
    return p;
  };
  __bf16* xn   = (__bf16*)carve((size_t)MM * EE * 2);        //  8 MiB (reused for LN2)
  __bf16* wqkv = (__bf16*)carve((size_t)EE * 3 * EE * 2);    //  6 MiB
  float*  bqkv = (float*) carve((size_t)3 * EE * 4);
  __bf16* wo_b = (__bf16*)carve((size_t)EE * EE * 2);        //  2 MiB
  __bf16* w1_b = (__bf16*)carve((size_t)EE * DFFN * 2);      //  8 MiB
  __bf16* w2_b = (__bf16*)carve((size_t)DFFN * EE * 2);      //  8 MiB
  float*  x1   = (float*) carve((size_t)MM * EE * 4);        // 16 MiB
  __bf16* qkvb = (__bf16*)carve((size_t)MM * 3 * EE * 2);    // 24 MiB
  __bf16* obuf = (__bf16*)carve((size_t)MM * EE * 2);        //  8 MiB (contiguous after qkvb)
  __bf16* hbuf = qkvb;  // alias: qkv(24MiB)+obuf(8MiB) = 32MiB, both dead before FFN1

  // pack weights/bias to bf16
  pack_qkv_w_kernel<<<(3*HH*EE*DHH)/256, 256, 0, stream>>>(Wq, Wk, Wv, wqkv);
  pack_qkv_b_kernel<<<(3*EE)/256, 256, 0, stream>>>(bq, bk, bv, bqkv);
  cast_bf16_kernel<<<(EE*EE)/256,   256, 0, stream>>>(Wo, wo_b, EE*EE);
  cast_bf16_kernel<<<(EE*DFFN)/256, 256, 0, stream>>>(W1, w1_b, EE*DFFN);
  cast_bf16_kernel<<<(DFFN*EE)/256, 256, 0, stream>>>(W2, w2_b, DFFN*EE);

  // LN1
  ln_bf16_kernel<<<MM, 256, 0, stream>>>(x, g1, be1, xn);
  // QKV projection: [4096,1024] @ [1024,3072] (+bias) -> bf16
  gemm_bf16_kernel<0><<<dim3(3*EE/BN, MM/BM), 256, 0, stream>>>(
      xn, wqkv, bqkv, nullptr, qkvb, nullptr, MM, 3*EE, EE);
  // causal multi-head attention
  flash_attn_kernel<<<dim3(TT/QROWS, BB*HH), 256, 0, stream>>>(qkvb, obuf);
  // out-proj + residual: x1 = x + O @ Wo + bo  (fp32)
  gemm_bf16_kernel<1><<<dim3(EE/BN, MM/BM), 256, 0, stream>>>(
      obuf, wo_b, bo, x, nullptr, x1, MM, EE, EE);
  // LN2
  ln_bf16_kernel<<<MM, 256, 0, stream>>>(x1, g2, be2, xn);
  // FFN1: relu(xn @ W1 + b1) -> bf16
  gemm_bf16_kernel<2><<<dim3(DFFN/BN, MM/BM), 256, 0, stream>>>(
      xn, w1_b, b1, nullptr, hbuf, nullptr, MM, DFFN, EE);
  // FFN2: out = x1 + h @ W2 + b2  (fp32)
  gemm_bf16_kernel<1><<<dim3(EE/BN, MM/BM), 256, 0, stream>>>(
      hbuf, w2_b, b2, x1, nullptr, out, MM, EE, DFFN);
}
